// Lofi2LofiModel_644245094505
// MI455X (gfx1250) — matrix operand order repaired and checked by reading, not compiled, not run on hardware
//
#include <hip/hip_runtime.h>
#include <hip/hip_bf16.h>

// ---------------------------------------------------------------- constants
#define kH       512
#define kB       128
#define kG       2048          // 4*H
#define kH2      128
#define kTC      64            // chord steps
#define kNPC     8
#define kTM      512           // melody steps (kTC*kNPC)
#define kCHORD_N 9
#define kMEL_N   130
#define kKEYS    12
#define kMODES   6
#define kFEAT    2688          // 4*H + 5*H2
#define kWAVES   8
#define kNT      4             // 16-wide N-tiles per wave (16x64 strip)

typedef __attribute__((ext_vector_type(16))) __bf16    v16bf;
typedef __attribute__((ext_vector_type(8)))  float     v8f;
typedef __attribute__((ext_vector_type(8)))  unsigned  v8u;

// pack two f32 into one dword holding two bf16 (truncate): one v_perm_b32
__device__ __forceinline__ unsigned pack2bf(float lo, float hi) {
  return __builtin_amdgcn_perm(__builtin_bit_cast(unsigned, hi),
                               __builtin_bit_cast(unsigned, lo), 0x07060302u);
}

__device__ __forceinline__ float sigf(float x) { return 1.0f / (1.0f + __expf(-x)); }

// ------------------------------------------------------------ WMMA GEMM
// C[M,N] (+)= A[M,K] @ W[N,K]^T (+ bias[n]) (ReLU).  Requires K % 32 == 0,
// lda/ldw % 4 == 0 (true for every call site).  One wave owns a 16x64 C strip:
// one A fragment is reused across 4 B fragments -> 4 v_wmma per 32-K chunk.
// Out-of-range B rows are CLAMPED (not masked): they only feed columns n>=N
// which are never stored, so the K-loop is branch-free and fully vectorized.
__global__ __launch_bounds__(kWAVES * 32)
void wmma_gemm(const float* __restrict__ A, int lda,
               const float* __restrict__ W, int ldw,
               float* __restrict__ C, int ldc,
               const float* __restrict__ bias,
               int M, int N, int K, int accum, int relu,
               int tiles_m, int tiles_ng) {
  int wave = (int)(threadIdx.x >> 5);
  int lane = (int)(threadIdx.x & 31);
  int tile = (int)blockIdx.x * kWAVES + wave;
  if (tile >= tiles_m * tiles_ng) return;
  int tm = tile / tiles_ng;
  int tg = tile - tm * tiles_ng;
  int m0 = tm * 16;
  int n0 = tg * (16 * kNT);

  int half = lane >> 4;        // 0 or 1
  int r    = lane & 15;

  int arow = m0 + r;
  if (arow >= M) arow = M - 1;                    // clamp (M mult of 16 anyway)
  const float* Ap = A + (size_t)arow * lda;

  const float* Wp[kNT];
#pragma unroll
  for (int j = 0; j < kNT; ++j) {
    int wrow = n0 + j * 16 + r;
    if (wrow >= N) wrow = N - 1;                  // clamp: cols n>=N never stored
    Wp[j] = W + (size_t)wrow * ldw;
  }

  v8f acc[kNT];
#pragma unroll
  for (int j = 0; j < kNT; ++j) acc[j] = (v8f){0.f,0.f,0.f,0.f,0.f,0.f,0.f,0.f};

  for (int kc = 0; kc < K; kc += 32) {
    __builtin_prefetch(Ap + kc + 32, 0, 1);       // speculative: safe past end

    // ---- A fragment: elems 0..15 <- K = kc + half*16 + i ----
    const float4 aL = *(const float4*)(Ap + kc + half * 16);
    const float4 aH = *(const float4*)(Ap + kc + half * 16 + 4);
    const float4 bL = *(const float4*)(Ap + kc + half * 16 + 8);
    const float4 bH = *(const float4*)(Ap + kc + half * 16 + 12);
    v8u au;
    au[0] = pack2bf(aL.x, aL.y);  au[1] = pack2bf(aL.z, aL.w);
    au[2] = pack2bf(aH.x, aH.y);  au[3] = pack2bf(aH.z, aH.w);
    au[4] = pack2bf(bL.x, bL.y);  au[5] = pack2bf(bL.z, bL.w);
    au[6] = pack2bf(bH.x, bH.y);  au[7] = pack2bf(bH.z, bH.w);
    v16bf afrag = __builtin_bit_cast(v16bf, au);

#pragma unroll
    for (int j = 0; j < kNT; ++j) {
      // ---- B fragment: elems 0..15 <- K = kc + half*16 + i ----
      const float* wp = Wp[j] + kc + half * 16;
      const float4 w0 = *(const float4*)(wp + 0);
      const float4 w1 = *(const float4*)(wp + 4);
      const float4 w2 = *(const float4*)(wp + 8);
      const float4 w3 = *(const float4*)(wp + 12);
      v8u bu;
      bu[0] = pack2bf(w0.x, w0.y);  bu[1] = pack2bf(w0.z, w0.w);
      bu[2] = pack2bf(w1.x, w1.y);  bu[3] = pack2bf(w1.z, w1.w);
      bu[4] = pack2bf(w2.x, w2.y);  bu[5] = pack2bf(w2.z, w2.w);
      bu[6] = pack2bf(w3.x, w3.y);  bu[7] = pack2bf(w3.z, w3.w);
      v16bf bfrag = __builtin_bit_cast(v16bf, bu);

      acc[j] = __builtin_amdgcn_wmma_f32_16x16x32_bf16(false, afrag, false, bfrag,
                                                       (short)0, acc[j], false, false);
    }
  }

#pragma unroll
  for (int j = 0; j < kNT; ++j) {
    int n = n0 + j * 16 + r;
    if (n < N) {
      float bv = bias ? bias[n] : 0.f;
#pragma unroll
      for (int v = 0; v < 8; ++v) {
        int m = m0 + half * 8 + v;                // D layout: half selects M+8
        if (m < M) {
          size_t off = (size_t)m * ldc + n;
          float val = acc[j][v] + bv;
          if (accum) val += C[off];
          if (relu)  val = fmaxf(val, 0.f);
          C[off] = val;
        }
      }
    }
  }
}

// ------------------------------------------------------------ elementwise
__global__ void lstm_gate(const float* __restrict__ gates, float* __restrict__ h,
                          float* __restrict__ c, float* __restrict__ y, int ldy) {
  int idx = (int)(blockIdx.x * blockDim.x + threadIdx.x);
  if (idx >= kB * kH) return;
  int b = idx / kH, j = idx - b * kH;
  const float* g = gates + (size_t)b * kG;
  float ig = g[j], fg = g[kH + j], gg = g[2 * kH + j], og = g[3 * kH + j];
  float cn = sigf(fg) * c[idx] + sigf(ig) * tanhf(gg);
  float hn = sigf(og) * tanhf(cn);
  c[idx] = cn;
  h[idx] = hn;
  if (y) y[(size_t)b * ldy + j] = hn;
}

__global__ void embed_seq(const float* __restrict__ emb, const int* __restrict__ idx,
                          float* __restrict__ out, int T) {
  long i = (long)blockIdx.x * blockDim.x + threadIdx.x;
  long total = (long)T * kB * kH;
  if (i >= total) return;
  int j = (int)(i % kH);
  long tb = i / kH;
  int b = (int)(tb % kB);
  int t = (int)(tb / kB);
  out[i] = emb[(size_t)idx[(size_t)b * T + t] * kH + j];   // out[t][b][j]
}

__global__ void gather_rows(const float* __restrict__ emb, const int* __restrict__ idx,
                            float* __restrict__ out, int D, int ldo) {
  int i = (int)(blockIdx.x * blockDim.x + threadIdx.x);
  if (i >= kB * D) return;
  int b = i / D, j = i - b * D;
  out[(size_t)b * ldo + j] = emb[(size_t)idx[b] * D + j];
}

__global__ void scalar_emb(const float* __restrict__ x, const float* __restrict__ W,
                           const float* __restrict__ bias, float* __restrict__ out,
                           int D, int ldo) {
  int i = (int)(blockIdx.x * blockDim.x + threadIdx.x);
  if (i >= kB * D) return;
  int b = i / D, j = i - b * D;
  out[(size_t)b * ldo + j] = x[b] * W[j] + bias[j];
}

__global__ void copy_rows(const float* __restrict__ src, int lds,
                          float* __restrict__ dst, int ldd, int D) {
  int i = (int)(blockIdx.x * blockDim.x + threadIdx.x);
  if (i >= kB * D) return;
  int b = i / D, j = i - b * D;
  dst[(size_t)b * ldd + j] = src[(size_t)b * lds + j];
}

__global__ void add_vec(const float* __restrict__ a, const float* __restrict__ b,
                        float* __restrict__ out, int n) {
  int i = (int)(blockIdx.x * blockDim.x + threadIdx.x);
  if (i < n) out[i] = a[i] + b[i];
}

__global__ void zero_kernel(float* __restrict__ p, int n) {
  int i = (int)(blockIdx.x * blockDim.x + threadIdx.x);
  if (i < n) p[i] = 0.f;
}

__global__ void argmax_embed(const float* __restrict__ logits, int ldl, int ncls,
                             const float* __restrict__ emb, float* __restrict__ out) {
  int b = (int)blockIdx.x;
  __shared__ int s_idx;
  if (threadIdx.x == 0) {
    const float* row = logits + (size_t)b * ldl;
    int best = 0;
    float bv = row[0];
    for (int k = 1; k < ncls; ++k) {
      float v = row[k];
      if (v > bv) { bv = v; best = k; }
    }
    s_idx = best;
  }
  __syncthreads();
  int ai = s_idx;
  for (int j = (int)threadIdx.x; j < kH; j += (int)blockDim.x)
    out[(size_t)b * kH + j] = emb[(size_t)ai * kH + j];
}

// single-block deterministic reduction: kl = 0.5*mean(mu^2 + exp(lv) - lv - 1)
__global__ void kl_reduce(const float* __restrict__ mu, const float* __restrict__ lv,
                          float* __restrict__ out, int n) {
  __shared__ float sm[256];
  float acc = 0.f;
  for (int i = (int)threadIdx.x; i < n; i += 256) {
    float m = mu[i], l = lv[i];
    acc += m * m + __expf(l) - l - 1.f;
  }
  sm[threadIdx.x] = acc;
  __syncthreads();
  for (int s = 128; s > 0; s >>= 1) {
    if ((int)threadIdx.x < s) sm[threadIdx.x] += sm[threadIdx.x + s];
    __syncthreads();
  }
  if (threadIdx.x == 0) out[0] = 0.5f * sm[0] / (float)n;
}

// ------------------------------------------------------------ host helpers
struct CellP { const float *Wih, *Whh, *bih, *bhh; int din; };

static inline void gemm(hipStream_t st, const float* A, int lda, const float* W, int ldw,
                        float* C, int ldc, const float* bias, int M, int N, int K,
                        int accum, int relu) {
  int tm = (M + 15) / 16;
  int tg = (N + 16 * kNT - 1) / (16 * kNT);
  int blocks = (tm * tg + kWAVES - 1) / kWAVES;
  wmma_gemm<<<blocks, kWAVES * 32, 0, st>>>(A, lda, W, ldw, C, ldc, bias, M, N, K,
                                            accum, relu, tm, tg);
}

static inline void zero_f(hipStream_t st, float* p, int n) {
  zero_kernel<<<(n + 255) / 256, 256, 0, st>>>(p, n);
}

static inline void lstm_step(hipStream_t st, const float* x, int ldx, const CellP& cl,
                             float* h, float* c, float* gates, float* y, int ldy) {
  gemm(st, x, ldx, cl.Wih, cl.din, gates, kG, cl.bih, kB, kG, cl.din, 0, 0);
  gemm(st, h, kH, cl.Whh, kH, gates, kG, cl.bhh, kB, kG, kH, 1, 0);
  lstm_gate<<<(kB * kH + 255) / 256, 256, 0, st>>>(gates, h, c, y, ldy);
}

static inline void run_dir(hipStream_t st, const float* xseq, int din, int T,
                           const CellP& cl, float* h, float* c, float* gates,
                           float* ybase, int yoff, int rev) {
  zero_f(st, h, kB * kH);
  zero_f(st, c, kB * kH);
  for (int s = 0; s < T; ++s) {
    int ti = rev ? (T - 1 - s) : s;
    const float* xt = xseq + (size_t)ti * kB * din;
    float* y = ybase ? (ybase + (size_t)ti * kB * (2 * kH) + yoff) : nullptr;
    lstm_step(st, xt, din, cl, h, c, gates, y, 2 * kH);
  }
}

// ------------------------------------------------------------ entry point
extern "C" void kernel_launch(void* const* d_in, const int* in_sizes, int n_in,
                              void* d_out, int out_size, void* d_ws, size_t ws_size,
                              hipStream_t stream) {
  (void)in_sizes; (void)n_in; (void)out_size; (void)ws_size;
  hipStream_t st = stream;
#define PF(i) ((const float*)d_in[i])
#define PI(i) ((const int*)d_in[i])

  // ---- JAX pytree-ordered inputs (top-level keys sorted) ----
  const float* cx_chords = PF(1);
  const float* cx_melody = PF(2);
  const int*   gt_chords = PI(3);
  const float* gt_energy = PF(4);
  const int*   gt_key = PI(5);
  const int*   gt_melodies = PI(6);
  const int*   gt_mode = PI(7);
  const float* gt_tempo = PF(8);
  const float* gt_valence = PF(9);
  const float* hx_chords = PF(10);
  const float* hx_melody = PF(11);

  // params (sorted keys; cells flatten Wih,Whh,bih,bhh; layers flatten b,f)
  CellP dec_ch  = {PF(13), PF(14), PF(15), PF(16), kH};
  const float *chdsW = PF(17), *chdsb = PF(18);
  const float *dec_chord_emb = PF(19);
  const float *cp1W = PF(20), *cp1b = PF(21), *cp2W = PF(22), *cp2b = PF(23);
  CellP dec_mel = {PF(24), PF(25), PF(26), PF(27), kH};
  const float *mdsW = PF(28), *mdsb = PF(29);
  const float *dec_mel_emb = PF(30);
  const float *mp1W = PF(31), *mp1b = PF(32), *mp2W = PF(33), *mp2b = PF(34);
  const float *downW = PF(35), *downb = PF(36);
  CellP ch_l0b = {PF(37), PF(38), PF(39), PF(40), kH};
  CellP ch_l0f = {PF(41), PF(42), PF(43), PF(44), kH};
  CellP ch_l1b = {PF(45), PF(46), PF(47), PF(48), 2 * kH};
  CellP ch_l1f = {PF(49), PF(50), PF(51), PF(52), 2 * kH};
  const float *enc_chord_emb = PF(53), *enc_mel_emb = PF(54);
  CellP me_l0b = {PF(55), PF(56), PF(57), PF(58), kH};
  CellP me_l0f = {PF(59), PF(60), PF(61), PF(62), kH};
  CellP me_l1b = {PF(63), PF(64), PF(65), PF(66), 2 * kH};
  CellP me_l1f = {PF(67), PF(68), PF(69), PF(70), 2 * kH};
  const float *enW = PF(71), *enb = PF(72);
  const float *eh1W = PF(73), *eh1b = PF(74), *eh2W = PF(75), *eh2b = PF(76);
  const float *key_emb = PF(77);
  const float *kh1W = PF(78), *kh1b = PF(79), *kh2W = PF(80), *kh2b = PF(81);
  const float *meanW = PF(82), *meanb = PF(83);
  const float *mode_emb = PF(84);
  const float *oh1W = PF(85), *oh1b = PF(86), *oh2W = PF(87), *oh2b = PF(88);
  const float *teW = PF(89), *teb = PF(90);
  const float *th1W = PF(91), *th1b = PF(92), *th2W = PF(93), *th2b = PF(94);
  const float *vaW = PF(95), *vab = PF(96);
  const float *vh1W = PF(97), *vh1b = PF(98), *vh2W = PF(99), *vh2b = PF(100);
  const float *varW = PF(101), *varb = PF(102);

  // ---- output layout (tuple flattened) ----
  float* out = (float*)d_out;
  const size_t OUT_CHORDS = 0;
  const size_t OUT_NOTES  = (size_t)kB * kTC * kCHORD_N;            // 73728
  const size_t OUT_TEMPO  = OUT_NOTES + (size_t)kB * kTM * kMEL_N;  // 8593408
  const size_t OUT_KEY    = OUT_TEMPO + kB;
  const size_t OUT_MODE   = OUT_KEY + (size_t)kB * kKEYS;
  const size_t OUT_VAL    = OUT_MODE + (size_t)kB * kMODES;
  const size_t OUT_EN     = OUT_VAL + kB;
  const size_t OUT_KL     = OUT_EN + kB;

  // ---- workspace carve-up ----
  float* ws = (float*)d_ws;
  size_t o = 0;
  auto alloc = [&](size_t n) { float* p = ws + o; o += n; return p; };
  float* ch_x    = alloc((size_t)kTC * kB * kH);
  float* ch_cat  = alloc((size_t)kTC * kB * 2 * kH);
  float* mel_x   = alloc((size_t)kTM * kB * kH);
  float* mel_cat = alloc((size_t)kTM * kB * 2 * kH);
  float* gates   = alloc((size_t)kB * kG);
  float* hbuf    = alloc((size_t)kB * kH);
  float* cbuf    = alloc((size_t)kB * kH);
  float* hf_ch   = alloc((size_t)kB * kH);
  float* hb_ch   = alloc((size_t)kB * kH);
  float* hf_mel  = alloc((size_t)kB * kH);
  float* hb_mel  = alloc((size_t)kB * kH);
  float* feats   = alloc((size_t)kB * kFEAT);
  float* henc    = alloc((size_t)kB * kH);
  float* mu      = alloc((size_t)kB * kH);     // z == mu (eval mode)
  float* lv      = alloc((size_t)kB * kH);
  float* htmp    = alloc((size_t)kB * kH2);
  float* hc      = alloc((size_t)kB * kH);
  float* cc      = alloc((size_t)kB * kH);
  float* hm      = alloc((size_t)kB * kH);
  float* cm      = alloc((size_t)kB * kH);
  float* ce      = alloc((size_t)kB * kH);
  float* me      = alloc((size_t)kB * kH);
  float* mel_prev= alloc((size_t)kB * kH);
  float* emb_tmp = alloc((size_t)kB * kH);
  float* ptmp    = alloc((size_t)kB * kH);

  const int EW = 256;
#define GRID1(n) dim3(((n) + EW - 1) / EW), dim3(EW), 0, st

  // ================= Encoder =================
  {
    long n1 = (long)kTC * kB * kH;
    embed_seq<<<dim3((unsigned)((n1 + EW - 1) / EW)), dim3(EW), 0, st>>>(
        enc_chord_emb, gt_chords, ch_x, kTC);
    long n2 = (long)kTM * kB * kH;
    embed_seq<<<dim3((unsigned)((n2 + EW - 1) / EW)), dim3(EW), 0, st>>>(
        enc_mel_emb, gt_melodies, mel_x, kTM);
  }
  // chord biLSTM
  run_dir(st, ch_x, kH, kTC, ch_l0f, hbuf, cbuf, gates, ch_cat, 0, 0);
  run_dir(st, ch_x, kH, kTC, ch_l0b, hbuf, cbuf, gates, ch_cat, kH, 1);
  run_dir(st, ch_cat, 2 * kH, kTC, ch_l1f, hf_ch, cbuf, gates, nullptr, 0, 0);
  run_dir(st, ch_cat, 2 * kH, kTC, ch_l1b, hb_ch, cbuf, gates, nullptr, 0, 1);
  // melody biLSTM
  run_dir(st, mel_x, kH, kTM, me_l0f, hbuf, cbuf, gates, mel_cat, 0, 0);
  run_dir(st, mel_x, kH, kTM, me_l0b, hbuf, cbuf, gates, mel_cat, kH, 1);
  run_dir(st, mel_cat, 2 * kH, kTM, me_l1f, hf_mel, cbuf, gates, nullptr, 0, 0);
  run_dir(st, mel_cat, 2 * kH, kTM, me_l1b, hb_mel, cbuf, gates, nullptr, 0, 1);

  // feats = [h_ch_b, h_ch_f, h_mel_b, h_mel_f, tempo, key, mode, valence, energy]
  copy_rows<<<GRID1(kB * kH)>>>(hb_ch, kH, feats + 0, kFEAT, kH);
  copy_rows<<<GRID1(kB * kH)>>>(hf_ch, kH, feats + kH, kFEAT, kH);
  copy_rows<<<GRID1(kB * kH)>>>(hb_mel, kH, feats + 2 * kH, kFEAT, kH);
  copy_rows<<<GRID1(kB * kH)>>>(hf_mel, kH, feats + 3 * kH, kFEAT, kH);
  scalar_emb<<<GRID1(kB * kH2)>>>(gt_tempo, teW, teb, feats + 4 * kH, kH2, kFEAT);
  gather_rows<<<GRID1(kB * kH2)>>>(key_emb, gt_key, feats + 4 * kH + kH2, kH2, kFEAT);
  gather_rows<<<GRID1(kB * kH2)>>>(mode_emb, gt_mode, feats + 4 * kH + 2 * kH2, kH2, kFEAT);
  scalar_emb<<<GRID1(kB * kH2)>>>(gt_valence, vaW, vab, feats + 4 * kH + 3 * kH2, kH2, kFEAT);
  scalar_emb<<<GRID1(kB * kH2)>>>(gt_energy, enW, enb, feats + 4 * kH + 4 * kH2, kH2, kFEAT);

  gemm(st, feats, kFEAT, downW, kFEAT, henc, kH, downb, kB, kH, kFEAT, 0, 0);
  gemm(st, henc, kH, meanW, kH, mu, kH, meanb, kB, kH, kH, 0, 0);
  gemm(st, henc, kH, varW, kH, lv, kH, varb, kB, kH, kH, 0, 0);
  kl_reduce<<<dim3(1), dim3(256), 0, st>>>(mu, lv, out + OUT_KL, kB * kH);

  // ================= Heads =================
  gemm(st, mu, kH, th1W, kH, htmp, kH2, th1b, kB, kH2, kH, 0, 1);
  gemm(st, htmp, kH2, th2W, kH2, out + OUT_TEMPO, 1, th2b, kB, 1, kH2, 0, 0);
  gemm(st, mu, kH, kh1W, kH, htmp, kH2, kh1b, kB, kH2, kH, 0, 1);
  gemm(st, htmp, kH2, kh2W, kH2, out + OUT_KEY, kKEYS, kh2b, kB, kKEYS, kH2, 0, 0);
  gemm(st, mu, kH, oh1W, kH, htmp, kH2, oh1b, kB, kH2, kH, 0, 1);
  gemm(st, htmp, kH2, oh2W, kH2, out + OUT_MODE, kMODES, oh2b, kB, kMODES, kH2, 0, 0);
  gemm(st, mu, kH, vh1W, kH, htmp, kH2, vh1b, kB, kH2, kH, 0, 1);
  gemm(st, htmp, kH2, vh2W, kH2, out + OUT_VAL, 1, vh2b, kB, 1, kH2, 0, 0);
  gemm(st, mu, kH, eh1W, kH, htmp, kH2, eh1b, kB, kH2, kH, 0, 1);
  gemm(st, htmp, kH2, eh2W, kH2, out + OUT_EN, 1, eh2b, kB, 1, kH2, 0, 0);

  // ================= Autoregressive decoder =================
  copy_rows<<<GRID1(kB * kH)>>>(hx_chords, kH, hc, kH, kH);
  copy_rows<<<GRID1(kB * kH)>>>(cx_chords, kH, cc, kH, kH);
  copy_rows<<<GRID1(kB * kH)>>>(hx_melody, kH, hm, kH, kH);
  copy_rows<<<GRID1(kB * kH)>>>(cx_melody, kH, cm, kH, kH);
  zero_f(st, mel_prev, kB * kH);

  const int ldc_ch = kTC * kCHORD_N;   // 576  (row stride of [B,64,9])
  const int ldc_no = kTM * kMEL_N;     // 66560 (row stride of [B,512,130])

  for (int t = 0; t < kTC; ++t) {
    const float* ch_in = (t == 0) ? mu : ce;
    lstm_step(st, ch_in, kH, dec_ch, hc, cc, gates, nullptr, 0);

    // chord logits -> directly into d_out at [b, t, :]
    float* cp_out = out + OUT_CHORDS + (size_t)t * kCHORD_N;
    gemm(st, hc, kH, cp1W, kH, ptmp, kH, cp1b, kB, kH, kH, 0, 1);
    gemm(st, ptmp, kH, cp2W, kH, cp_out, ldc_ch, cp2b, kB, kCHORD_N, kH, 0, 0);
    argmax_embed<<<dim3(kB), dim3(256), 0, st>>>(cp_out, ldc_ch, kCHORD_N,
                                                 dec_chord_emb, emb_tmp);
    // ce = [emb, z] @ chord_ds.W^T + b   (W is [512,1024], split K)
    gemm(st, emb_tmp, kH, chdsW, 2 * kH, ce, kH, chdsb, kB, kH, kH, 0, 0);
    gemm(st, mu, kH, chdsW + kH, 2 * kH, ce, kH, nullptr, kB, kH, kH, 1, 0);

    // mel_in = mel_emb + ce
    add_vec<<<GRID1(kB * kH)>>>(mel_prev, ce, me, kB * kH);

    for (int n = 0; n < kNPC; ++n) {
      lstm_step(st, me, kH, dec_mel, hm, cm, gates, nullptr, 0);
      float* mp_out = out + OUT_NOTES + (size_t)(t * kNPC + n) * kMEL_N;
      gemm(st, hm, kH, mp1W, kH, ptmp, kH, mp1b, kB, kH, kH, 0, 1);
      gemm(st, ptmp, kH, mp2W, kH, mp_out, ldc_no, mp2b, kB, kMEL_N, kH, 0, 0);
      argmax_embed<<<dim3(kB), dim3(256), 0, st>>>(mp_out, ldc_no, kMEL_N,
                                                   dec_mel_emb, emb_tmp);
      // me = [emb, ce, z] @ mel_ds.W^T + b  (W is [512,1536], split K into 3)
      gemm(st, emb_tmp, kH, mdsW, 3 * kH, me, kH, mdsb, kB, kH, kH, 0, 0);
      gemm(st, ce, kH, mdsW + kH, 3 * kH, me, kH, nullptr, kB, kH, kH, 1, 0);
      gemm(st, mu, kH, mdsW + 2 * kH, 3 * kH, me, kH, nullptr, kB, kH, kH, 1, 0);
    }
    copy_rows<<<GRID1(kB * kH)>>>(me, kH, mel_prev, kH, kH);  // carry mel_emb
  }
#undef GRID1
#undef PF
#undef PI
}
